// SparseMOE_47244640256432
// MI455X (gfx1250) — matrix operand order repaired and checked by reading, not compiled, not run on hardware
//
#include <hip/hip_runtime.h>
#include <hip/hip_bf16.h>

#define B_  4
#define S_  4096
#define D_  1024
#define E_  8
#define T_  (B_ * S_)          // 16384 tokens

typedef __attribute__((ext_vector_type(16))) __bf16 v16bf;
typedef __attribute__((ext_vector_type(8)))  __bf16 v8bf;
typedef __attribute__((ext_vector_type(8)))  float  v8f;
typedef int v4i_ __attribute__((vector_size(16)));

#if defined(__HIP_DEVICE_COMPILE__) && __has_builtin(__builtin_amdgcn_global_load_async_to_lds_b128)
#define HAVE_ASYNC 1
#else
#define HAVE_ASYNC 0
#endif

#if HAVE_ASYNC
#define ASYNC_CP_B128(dst, src, off)                                          \
    __builtin_amdgcn_global_load_async_to_lds_b128(                           \
        (__attribute__((address_space(1))) v4i_*)(v4i_*)(src),                \
        (__attribute__((address_space(3))) v4i_*)(v4i_*)(dst), (off), 0)
static __device__ __forceinline__ void wait_async0() {
#if __has_builtin(__builtin_amdgcn_s_wait_asynccnt)
    __builtin_amdgcn_s_wait_asynccnt(0);
#else
    asm volatile("s_wait_asynccnt 0" ::: "memory");
#endif
}
#endif

// ---------------------------------------------------------------------------
// Kernel 1: x fp32 -> bf16
// ---------------------------------------------------------------------------
__global__ void k_convert_x(const float* __restrict__ x, __bf16* __restrict__ xb) {
    size_t i = (size_t)blockIdx.x * blockDim.x + threadIdx.x;   // one v8 per thread
    const float4* x4 = (const float4*)x;
    float4 a = x4[2 * i];
    float4 b = x4[2 * i + 1];
    v8bf o;
    o[0] = (__bf16)a.x; o[1] = (__bf16)a.y; o[2] = (__bf16)a.z; o[3] = (__bf16)a.w;
    o[4] = (__bf16)b.x; o[5] = (__bf16)b.y; o[6] = (__bf16)b.z; o[7] = (__bf16)b.w;
    *(v8bf*)(xb + i * 8) = o;
}

// ---------------------------------------------------------------------------
// Kernel 2: expert_w [E][N][K] fp32 -> wt [E][K][N] bf16 (transposed)
// ---------------------------------------------------------------------------
__global__ void k_transpose_w(const float* __restrict__ w, __bf16* __restrict__ wt) {
    __shared__ float tile[32][33];
    const int e  = blockIdx.z;
    const float* we  = w  + (size_t)e * D_ * D_;
    __bf16*      wte = wt + (size_t)e * D_ * D_;
    const int c0 = blockIdx.x * 32;   // K block in source
    const int r0 = blockIdx.y * 32;   // N block in source
#pragma unroll
    for (int p = 0; p < 4; ++p) {
        int r = threadIdx.y + p * 8;
        tile[r][threadIdx.x] = we[(size_t)(r0 + r) * D_ + c0 + threadIdx.x];
    }
    __syncthreads();
#pragma unroll
    for (int p = 0; p < 4; ++p) {
        int r = threadIdx.y + p * 8;  // row within K block of dest
        wte[(size_t)(c0 + r) * D_ + r0 + threadIdx.x] = (__bf16)tile[threadIdx.x][r];
    }
}

// ---------------------------------------------------------------------------
// Kernel 3: router — logits, softmax, top-2, renorm; build per-expert lists
// One wave (32 lanes) per token.
// ---------------------------------------------------------------------------
__global__ void k_router(const float* __restrict__ x, const float* __restrict__ gw,
                         const float* __restrict__ gb, float* __restrict__ logits_out,
                         int* __restrict__ cnt, int* __restrict__ tok,
                         float* __restrict__ wgt) {
    const int wave = threadIdx.x >> 5, lane = threadIdx.x & 31;
    const int t = blockIdx.x * 8 + wave;
    const float4* xr = (const float4*)(x + (size_t)t * D_);
    const float4* g4 = (const float4*)gw;

    float acc[E_];
#pragma unroll
    for (int e = 0; e < E_; ++e) acc[e] = 0.f;
#pragma unroll
    for (int q = 0; q < 8; ++q) {
        float4 xv = xr[lane + q * 32];
#pragma unroll
        for (int e = 0; e < E_; ++e) {
            float4 gv = g4[e * 256 + lane + q * 32];
            acc[e] += xv.x * gv.x + xv.y * gv.y + xv.z * gv.z + xv.w * gv.w;
        }
    }
#pragma unroll
    for (int off = 16; off >= 1; off >>= 1)
#pragma unroll
        for (int e = 0; e < E_; ++e) acc[e] += __shfl_xor(acc[e], off, 32);

    float logit[E_];
#pragma unroll
    for (int e = 0; e < E_; ++e) logit[e] = acc[e] + gb[e];

    if (lane < E_) logits_out[(size_t)t * E_ + lane] = logit[lane];

    if (lane == 0) {
        int e1 = 0;
#pragma unroll
        for (int e = 1; e < E_; ++e) if (logit[e] > logit[e1]) e1 = e;
        int e2 = (e1 == 0) ? 1 : 0;
#pragma unroll
        for (int e = 0; e < E_; ++e)
            if (e != e1 && logit[e] > logit[e2]) e2 = e;
        // softmax denominator cancels in the top-2 renormalization
        float ex2 = __expf(logit[e2] - logit[e1]);   // exp1 == 1
        float inv = 1.0f / (1.0f + ex2);
        float w1 = inv, w2 = ex2 * inv;
        int p1 = atomicAdd(&cnt[e1], 1);
        tok[e1 * T_ + p1] = t;  wgt[e1 * T_ + p1] = w1;
        int p2 = atomicAdd(&cnt[e2], 1);
        tok[e2 * T_ + p2] = t;  wgt[e2 * T_ + p2] = w2;
    }
}

// ---------------------------------------------------------------------------
// Kernel 4: grouped expert GEMM with WMMA bf16 + async double-buffered staging
//   Y[t, n] = sum_k xb[t,k] * wt_e[k,n];  out[t,n] += w_t * (Y + bias_n)
// Block tile 128(M) x 128(N), 8 waves (256 thr), wave tile 32x64.
// ---------------------------------------------------------------------------
#define LDA 56     // bf16 pitch of A tile (112B rows: 16B aligned, conflict-free)
#define LDB 136    // bf16 pitch of B tile (272B rows)
#define ASZ (128 * LDA)
#define BSZ (32 * LDB)

__global__ __launch_bounds__(256)
void k_expert_gemm(const __bf16* __restrict__ xb, const __bf16* __restrict__ wt,
                   const float* __restrict__ eb, const int* __restrict__ cnt,
                   const int* __restrict__ tok, const float* __restrict__ wgt,
                   float* __restrict__ out) {
    const int e    = blockIdx.z;
    const int mblk = blockIdx.y;
    const int n0   = blockIdx.x * 128;
    const int count = cnt[e];
    if (mblk * 128 >= count) return;

#if HAVE_ASYNC
    __shared__ __bf16 sA[2 * ASZ];
    __shared__ __bf16 sB[2 * BSZ];
#else
    __shared__ __bf16 sA[ASZ];
    __shared__ __bf16 sB[BSZ];
#endif
    __shared__ int    sTok[128];
    __shared__ float  sWgt[128];
    __shared__ float  sBias[128];

    const int tid  = threadIdx.x;
    const int wave = tid >> 5, lane = tid & 31;

    if (tid < 128) {
        int row = mblk * 128 + tid;
        int valid = row < count;
        sTok[tid] = valid ? tok[e * T_ + row] : 0;
        sWgt[tid] = valid ? wgt[e * T_ + row] : 0.f;
    } else {
        sBias[tid - 128] = eb[(size_t)e * D_ + n0 + (tid - 128)];
    }

    const __bf16* wte = wt + (size_t)e * D_ * D_;

    // Per-thread staging geometry: each thread moves 32 contiguous bytes of
    // one A row and 32 contiguous bytes of one B row per K step.
    const int aRow = tid >> 1;            // 0..127
    const int aOff = (tid & 1) * 16;      // element offset 0 / 16
    const int bRow = tid >> 3;            // 0..31
    const int bOff = (tid & 7) * 16;      // element offset 0..112
    const int aTokRow = mblk * 128 + aRow;
    const int aTok = (aTokRow < count) ? tok[e * T_ + aTokRow] : 0;
    const __bf16* aSrc = xb  + (size_t)aTok * D_ + aOff;            // + kbase
    const __bf16* bSrc = wte + (size_t)bRow * D_ + n0 + bOff;       // + kbase*D
    __bf16* aDst = sA + aRow * LDA + aOff;                          // + buf*ASZ
    __bf16* bDst = sB + bRow * LDB + bOff;                          // + buf*BSZ

    const int wm0 = (wave & 3) * 32;   // wave M offset in block tile
    const int wn0 = (wave >> 2) * 64;  // wave N offset in block tile

    v8f acc[2][4];
#pragma unroll
    for (int i = 0; i < 2; ++i)
#pragma unroll
        for (int j = 0; j < 4; ++j) acc[i][j] = (v8f){0.f,0.f,0.f,0.f,0.f,0.f,0.f,0.f};

    const int k0   = (lane >> 4) * 8;    // A fragment K sub-offset per ISA layout
    const int mrow = lane & 15;

    auto compute_tiles = [&](const __bf16* cA, const __bf16* cB) {
        v16bf afr[2], bfr[4];
#pragma unroll
        for (int i = 0; i < 2; ++i) {
            const __bf16* p = cA + (wm0 + i * 16 + mrow) * LDA;
            v8bf lo = *(const v8bf*)(p + k0);
            v8bf hi = *(const v8bf*)(p + k0 + 16);
            afr[i] = __builtin_shufflevector(lo, hi, 0,1,2,3,4,5,6,7,
                                                     8,9,10,11,12,13,14,15);
        }
#pragma unroll
        for (int j = 0; j < 4; ++j) {
            const __bf16* p = cB + wn0 + j * 16 + mrow;   // lane%16 == N column
#pragma unroll
            for (int s = 0; s < 16; ++s)
                bfr[j][s] = p[(k0 + s + (s >= 8 ? 8 : 0)) * LDB];
        }
#pragma unroll
        for (int i = 0; i < 2; ++i)
#pragma unroll
            for (int j = 0; j < 4; ++j)
                acc[i][j] = __builtin_amdgcn_wmma_f32_16x16x32_bf16(
                    false, afr[i], false, bfr[j], (short)0, acc[i][j], false, false);
    };

#if HAVE_ASYNC
    // -------- async double-buffered pipeline: one barrier per K step --------
    {   // prologue: stage tile 0 into buffer 0
        ASYNC_CP_B128(aDst, aSrc, 0);
        ASYNC_CP_B128(aDst, aSrc, 16);
        ASYNC_CP_B128(bDst, bSrc, 0);
        ASYNC_CP_B128(bDst, bSrc, 16);
    }
    for (int kk = 0; kk < D_ / 32; ++kk) {
        const int cur = kk & 1;
        wait_async0();            // my writes to buf[cur] have landed
        __syncthreads();          // everyone's writes landed; prev compute done
        if (kk + 1 < D_ / 32) {   // stage next tile into the other buffer
            const int kb = (kk + 1) * 32;
            const __bf16* aS = aSrc + kb;
            const __bf16* bS = bSrc + (size_t)kb * D_;
            __bf16* aD = aDst + (1 - cur) * ASZ;
            __bf16* bD = bDst + (1 - cur) * BSZ;
            ASYNC_CP_B128(aD, aS, 0);
            ASYNC_CP_B128(aD, aS, 16);
            ASYNC_CP_B128(bD, bS, 0);
            ASYNC_CP_B128(bD, bS, 16);
        }
        compute_tiles(sA + cur * ASZ, sB + cur * BSZ);
    }
#else
    // -------- fallback: synchronous single-buffer staging --------
    for (int kk = 0; kk < D_ / 32; ++kk) {
        const int kb = kk * 32;
        const __bf16* aS = aSrc + kb;
        const __bf16* bS = bSrc + (size_t)kb * D_;
        v8bf a0 = *(const v8bf*)(aS);
        v8bf a1 = *(const v8bf*)(aS + 8);
        v8bf b0 = *(const v8bf*)(bS);
        v8bf b1 = *(const v8bf*)(bS + 8);
        __syncthreads();          // previous compute finished reading LDS
        *(v8bf*)(aDst)     = a0;
        *(v8bf*)(aDst + 8) = a1;
        *(v8bf*)(bDst)     = b0;
        *(v8bf*)(bDst + 8) = b1;
        __syncthreads();
        compute_tiles(sA, sB);
    }
#endif

    // ---- scale by routing weight, add bias, accumulate ----
    const int ncol  = lane & 15;          // C/D: lane = N column
    const int mhalf = (lane >> 4) * 8;    // lanes 16-31 hold rows M+8
#pragma unroll
    for (int i = 0; i < 2; ++i) {
#pragma unroll
        for (int j = 0; j < 4; ++j) {
            int nloc  = wn0 + j * 16 + ncol;
            float bias = sBias[nloc];
#pragma unroll
            for (int r = 0; r < 8; ++r) {
                int mloc = wm0 + i * 16 + mhalf + r;
                float w  = sWgt[mloc];
                if (w != 0.f) {
                    atomicAdd(out + (size_t)sTok[mloc] * D_ + n0 + nloc,
                              w * (acc[i][j][r] + bias));
                }
            }
        }
    }
}

// ---------------------------------------------------------------------------
extern "C" void kernel_launch(void* const* d_in, const int* in_sizes, int n_in,
                              void* d_out, int out_size, void* d_ws, size_t ws_size,
                              hipStream_t stream) {
    const float* x   = (const float*)d_in[0];   // [B,S,D]
    const float* gw  = (const float*)d_in[1];   // [E,D]
    const float* gb  = (const float*)d_in[2];   // [E]
    const float* ew  = (const float*)d_in[3];   // [E,D,D]
    const float* ebi = (const float*)d_in[4];   // [E,D]

    float* out    = (float*)d_out;                  // [T,D]
    float* logits = out + (size_t)T_ * D_;          // [T,E]

    char* ws = (char*)d_ws;
    __bf16* xb  = (__bf16*)ws;  ws += (size_t)T_ * D_ * 2;       // 32 MB
    __bf16* wtp = (__bf16*)ws;  ws += (size_t)E_ * D_ * D_ * 2;  // 16 MB
    int*    tok = (int*)ws;     ws += (size_t)E_ * T_ * 4;       // 512 KB
    float*  wgt = (float*)ws;   ws += (size_t)E_ * T_ * 4;       // 512 KB
    int*    cnt = (int*)ws;     ws += 256;

    (void)hipMemsetAsync(out, 0, (size_t)T_ * D_ * sizeof(float), stream);
    (void)hipMemsetAsync(cnt, 0, E_ * sizeof(int), stream);

    k_convert_x<<<(T_ * (D_ / 8)) / 256, 256, 0, stream>>>(x, xb);
    k_transpose_w<<<dim3(D_ / 32, D_ / 32, E_), dim3(32, 8), 0, stream>>>(ew, wtp);
    k_router<<<T_ / 8, 256, 0, stream>>>(x, gw, gb, logits, cnt, tok, wgt);
    k_expert_gemm<<<dim3(D_ / 128, T_ / 128, E_), 256, 0, stream>>>(
        xb, wtp, ebi, cnt, tok, wgt, out);
}
